// NUFNO2d_63410897158610
// MI455X (gfx1250) — matrix operand order repaired and checked by reading, not compiled, not run on hardware
//
#include <hip/hip_runtime.h>
#include <math.h>

// ---------------- CDNA5 WMMA types ----------------
typedef __attribute__((ext_vector_type(16))) _Float16 v16h;
typedef __attribute__((ext_vector_type(8)))  float    v8f;
typedef __attribute__((ext_vector_type(2)))  float    v2f;

#define WMMA_F16(a, b, c) \
  __builtin_amdgcn_wmma_f32_16x16x32_f16(false, (a), false, (b), (short)0, (c), false, false)
#define WMMA_F32(a, b, c) \
  __builtin_amdgcn_wmma_f32_16x16x4_f32(false, (a), false, (b), (short)0, (c), false, false)

// ---- async global->LDS staging (CDNA5 GLOBAL_LOAD_ASYNC_TO_LDS_B128, ASYNCcnt) ----
#if defined(__HIP_DEVICE_COMPILE__) && \
    __has_builtin(__builtin_amdgcn_global_load_async_to_lds_b128)
#define HAVE_ASYNC_LDS 1
typedef int v4i_b __attribute__((vector_size(16)));
typedef v4i_b __attribute__((address_space(1)))* g128_t;
typedef v4i_b __attribute__((address_space(3)))* l128_t;
__device__ __forceinline__ void async_copy16(const void* g, void* l) {
  __builtin_amdgcn_global_load_async_to_lds_b128((g128_t)g, (l128_t)l, 0, 0);
}
__device__ __forceinline__ void async_wait0() {
#if __has_builtin(__builtin_amdgcn_s_wait_asynccnt)
  __builtin_amdgcn_s_wait_asynccnt(0);
#else
  asm volatile("s_wait_asynccnt 0x0" ::: "memory");
#endif
}
#else
#define HAVE_ASYNC_LDS 0
#endif

__device__ __forceinline__ float gelu_exact(float x) {
  return 0.5f * x * (1.0f + erff(x * 0.70710678118654752f));
}

// ============================================================
// MODES=16, NCH=32, NSD=32, B=4, H=W=64, N=512; BS=128
// Planar spectral layouts:
//   tmpA / tmpD : (img, part{re,im}, 32, 64)  -> img*4096 + part*2048 + kk*64 + w
//   Xm / Ym     : (img, part{re,im}, 32, 16)  -> img*1024 + part*512  + kk*16 + k2
// ============================================================

// ---- factor-matrix tables: EA(64x64), EE(64x64), BB(128x32), BD(32x128) ----
__global__ void k_tables(float* __restrict__ EA, float* __restrict__ EE,
                         float* __restrict__ BB, float* __restrict__ BD) {
  int id = blockIdx.x * blockDim.x + threadIdx.x;
  if (id >= 4 * 4096) return;
  int sel = id >> 12, r = id & 4095;
  if (sel == 0) {               // EA[row][h]: rows [cos(k1 h); -sin(k1 h)]
    int row = r >> 6, h = r & 63;
    int kk = row & 31;
    int k1 = (kk < 16) ? kk : kk + 32;
    float ang = 6.28318530717958647f * (float)((k1 * h) & 63) * (1.0f / 64.0f);
    float s, c; __sincosf(ang, &s, &c);
    EA[r] = (row < 32) ? c : -s;
  } else if (sel == 1) {        // EE[h][kap]: [cos(k1 h); -sin(k1 h)] / 4096
    int h = r >> 6, kap = r & 63;
    int kk = kap & 31;
    int k1 = (kk < 16) ? kk : kk + 32;
    float ang = 6.28318530717958647f * (float)((k1 * h) & 63) * (1.0f / 64.0f);
    float s, c; __sincosf(ang, &s, &c);
    EE[r] = ((kap < 32) ? c : -s) * (1.0f / 4096.0f);
  } else if (sel == 2) {        // BB[kap<128][n<32]: Xr cols [C;S], Xi cols [-S;C]
    int kap = r >> 5, n = r & 31;
    int k2 = n & 15, w = kap & 63;
    float ang = 6.28318530717958647f * (float)((k2 * w) & 63) * (1.0f / 64.0f);
    float s, c; __sincosf(ang, &s, &c);
    BB[r] = (n < 16) ? ((kap < 64) ? c : s) : ((kap < 64) ? -s : c);
  } else {                      // BD[kap<32][n<128]: tr cols [aC;-aS], ti cols [aS;aC]
    int kap = r >> 7, n = r & 127;
    int k2 = kap & 15, w = n & 63;
    float a = (k2 == 0) ? 1.0f : 2.0f;
    float ang = 6.28318530717958647f * (float)((k2 * w) & 63) * (1.0f / 64.0f);
    float s, c; __sincosf(ang, &s, &c);
    BD[r] = (n < 64) ? ((kap < 16) ? a * c : -a * s) : ((kap < 16) ? a * s : a * c);
  }
}

// ---- fc0: concat(u, grid) @ fc0w + fc0b -> xg (b, c, h, w) ----
__global__ void k_fc0(const float* __restrict__ u, const float* __restrict__ fw,
                      const float* __restrict__ fb, float* __restrict__ xg) {
  int id = blockIdx.x * blockDim.x + threadIdx.x;
  if (id >= 4 * 4096) return;
  int b = id >> 12, hw = id & 4095, h = hw >> 6, w = hw & 63;
  float uv = u[id];
  float gx = (float)h * (1.0f / 63.0f);
  float gy = (float)w * (1.0f / 63.0f);
  for (int o = 0; o < 32; ++o) {
    float v = uv * fw[o] + gx * fw[32 + o] + gy * fw[64 + o] + fb[o];
    xg[((size_t)b * 32 + o) * 4096 + hw] = v;
  }
}

// ---- stage A (WMMA f32): tmpAp[img] = EA(64x64) @ x[img](64x64), planar re/im ----
__global__ void k_wA(const float* __restrict__ src, const float* __restrict__ EA,
                     float* __restrict__ outp) {
  int img = blockIdx.x, Mt = blockIdx.y, Nt = blockIdx.z;
  int lane = threadIdx.x & 31, m = lane & 15, hid = lane >> 4;
  const float* B = src + (size_t)img * 4096;
  v8f acc = {};
  for (int k0 = 0; k0 < 64; k0 += 4) {
    int k = k0 + hid * 2;
    v2f a, b;
    a[0] = EA[(Mt * 16 + m) * 64 + k];
    a[1] = EA[(Mt * 16 + m) * 64 + k + 1];
    b[0] = B[(size_t)k * 64 + Nt * 16 + m];
    b[1] = B[(size_t)(k + 1) * 64 + Nt * 16 + m];
    acc = WMMA_F32(a, b, acc);
  }
  for (int r = 0; r < 8; ++r) {
    int row = Mt * 16 + hid * 8 + r;
    outp[(size_t)img * 4096 + row * 64 + Nt * 16 + m] = acc[r];
  }
}

// ---- stage B (WMMA f32): Xm[img](32x32) = [Ar|Ai](32x128) @ BB(128x32) ----
__global__ void k_wB(const float* __restrict__ Ap, const float* __restrict__ BB,
                     float* __restrict__ Xmp) {
  int img = blockIdx.x, Mt = blockIdx.y, Nt = blockIdx.z;
  int lane = threadIdx.x & 31, m = lane & 15, hid = lane >> 4;
  const float* A0 = Ap + (size_t)img * 4096 + (Mt * 16 + m) * 64;
  v8f acc = {};
  for (int k0 = 0; k0 < 128; k0 += 4) {
    int k = k0 + hid * 2;
    v2f a, b;
    a[0] = (k < 64) ? A0[k] : A0[2048 + k - 64];
    a[1] = (k + 1 < 64) ? A0[k + 1] : A0[2048 + k - 63];
    b[0] = BB[(size_t)k * 32 + Nt * 16 + m];
    b[1] = BB[(size_t)(k + 1) * 32 + Nt * 16 + m];
    acc = WMMA_F32(a, b, acc);
  }
  for (int r = 0; r < 8; ++r) {
    int kk = Mt * 16 + hid * 8 + r, n = Nt * 16 + m;
    Xmp[(size_t)img * 1024 + (n >> 4) * 512 + kk * 16 + (n & 15)] = acc[r];
  }
}

// ---- channel mix (WMMA f32), per mode position:
//      [Yr|Yi](nb x 64) = [Xr|Xi](nb x 64) @ [[Wr,Wi],[-Wi,Wr]](64x64) ----
__global__ void k_mixw(const float* __restrict__ Xmp,
                       const float* __restrict__ w1r, const float* __restrict__ w1i,
                       const float* __restrict__ w2r, const float* __restrict__ w2i,
                       float* __restrict__ Ymp, int layer, int nb) {
  int pos = blockIdx.x, Mt = blockIdx.y, Nt = blockIdx.z;
  int kk = pos >> 4, k2 = pos & 15;
  int lane = threadIdx.x & 31, m = lane & 15, hid = lane >> 4;
  const float* wr; const float* wi; int mi;
  if (kk < 16) { wr = w1r; wi = w1i; mi = kk; }
  else         { wr = w2r; wi = w2i; mi = kk - 16; }
  size_t lb = (size_t)layer * 32 * 32 * 256;
  int brow = Mt * 16 + m;                // batch row for A
  int n = Nt * 16 + m;                   // output col: (po, o)
  int o = n & 31, po = n >> 5;
  v8f acc = {};
  for (int k0 = 0; k0 < 64; k0 += 4) {
    v2f a, b;
    for (int v = 0; v < 2; ++v) {
      int kap = k0 + hid * 2 + v;
      int i = kap & 31, pi = kap >> 5;
      a[v] = (brow < nb)
               ? Xmp[((size_t)brow * 32 + i) * 1024 + pi * 512 + kk * 16 + k2]
               : 0.0f;
      size_t widx = lb + (((size_t)i * 32 + o) * 16 + mi) * 16 + k2;
      float wrv = wr[widx], wiv = wi[widx];
      b[v] = (po == 0) ? ((pi == 0) ? wrv : -wiv)
                       : ((pi == 0) ? wiv :  wrv);
    }
    acc = WMMA_F32(a, b, acc);
  }
  for (int r = 0; r < 8; ++r) {
    int bo = Mt * 16 + hid * 8 + r;
    if (bo < nb)
      Ymp[((size_t)bo * 32 + o) * 1024 + po * 512 + kk * 16 + k2] = acc[r];
  }
}

// ---- stage D (WMMA f32): [tr|ti](32x128) = [Yr|Yi](32x32) @ BD(32x128) ----
__global__ void k_wD(const float* __restrict__ Ymp, const float* __restrict__ BD,
                     float* __restrict__ outp) {
  int img = blockIdx.x, Mt = blockIdx.y, Nt = blockIdx.z;
  int lane = threadIdx.x & 31, m = lane & 15, hid = lane >> 4;
  const float* A0 = Ymp + (size_t)img * 1024 + (Mt * 16 + m) * 16;
  v8f acc = {};
  for (int k0 = 0; k0 < 32; k0 += 4) {
    int k = k0 + hid * 2;
    v2f a, b;
    a[0] = A0[(k >> 4) * 512 + (k & 15)];
    a[1] = A0[((k + 1) >> 4) * 512 + ((k + 1) & 15)];
    b[0] = BD[(size_t)k * 128 + Nt * 16 + m];
    b[1] = BD[(size_t)(k + 1) * 128 + Nt * 16 + m];
    acc = WMMA_F32(a, b, acc);
  }
  for (int r = 0; r < 8; ++r) {
    int kk = Mt * 16 + hid * 8 + r, n = Nt * 16 + m;
    outp[(size_t)img * 4096 + (n >> 6) * 2048 + kk * 64 + (n & 63)] = acc[r];
  }
}

// ---- stage E (WMMA f32): y(64x64) = EE(64x64) @ [tr;ti](64x64) ----
__global__ void k_wE(const float* __restrict__ EE, const float* __restrict__ Dp,
                     float* __restrict__ yb) {
  int img = blockIdx.x, Mt = blockIdx.y, Nt = blockIdx.z;
  int lane = threadIdx.x & 31, m = lane & 15, hid = lane >> 4;
  const float* B0 = Dp + (size_t)img * 4096;
  v8f acc = {};
  for (int k0 = 0; k0 < 64; k0 += 4) {
    int k = k0 + hid * 2;
    v2f a, b;
    a[0] = EE[(Mt * 16 + m) * 64 + k];
    a[1] = EE[(Mt * 16 + m) * 64 + k + 1];
    b[0] = B0[(size_t)k * 64 + Nt * 16 + m];
    b[1] = B0[(size_t)(k + 1) * 64 + Nt * 16 + m];
    acc = WMMA_F32(a, b, acc);
  }
  for (int r = 0; r < 8; ++r) {
    int h = Mt * 16 + hid * 8 + r;
    yb[(size_t)img * 4096 + h * 64 + Nt * 16 + m] = acc[r];
  }
}

// ---- per-site 1x1 conv + bias + GELU, in place on x.
//      Weights staged into LDS via async global->LDS copy (ASYNCcnt). ----
__global__ void k_convact(const float* __restrict__ yb, float* __restrict__ x,
                          const float* __restrict__ cw, const float* __restrict__ cb,
                          int layer, int nb) {
  __shared__ float sW[1024];           // 32x32 conv weights
  __shared__ float sB[32];
  int tid = threadIdx.x;
  const float* wsrc = cw + (size_t)layer * 1024;
#if HAVE_ASYNC_LDS
  async_copy16(wsrc + tid * 4, &sW[tid * 4]);   // 256 lanes x 16B = 4KB
  async_wait0();
#else
  for (int i = tid; i < 1024; i += 256) sW[i] = wsrc[i];
#endif
  if (tid < 32) sB[tid] = cb[layer * 32 + tid];
  __syncthreads();

  int id = blockIdx.x * 256 + tid;
  if (id >= nb * 4096) return;
  int hw = id & 4095, b = id >> 12;
  float in[32];
  for (int i = 0; i < 32; ++i) in[i] = x[((size_t)b * 32 + i) * 4096 + hw];
  for (int o = 0; o < 32; ++o) {
    float v = yb[((size_t)b * 32 + o) * 4096 + hw] + sB[o];
    const float* wrow = &sW[o * 32];
    for (int i = 0; i < 32; ++i) v += wrow[i] * in[i];
    x[((size_t)b * 32 + o) * 4096 + hw] = gelu_exact(v);
  }
}

// ---- GEMM1 (WMMA f16): x^T (16384x32) @ fi0w (32x128), gelu -> t1 f16 ----
__global__ void k_gemm1(const float* __restrict__ xg, const float* __restrict__ fw,
                        const float* __restrict__ fb, _Float16* __restrict__ t1) {
  int Mt = blockIdx.x, Nt = blockIdx.y;
  int lane = threadIdx.x & 31, m = lane & 15, hid = lane >> 4;
  int p = Mt * 16 + m, b = p >> 12, hw = p & 4095;
  int n = Nt * 16 + m;
  v16h av, bv; v8f acc = {};
  for (int j = 0; j < 16; ++j) {
    int ka = hid * 16 + j;
    av[j] = (_Float16)xg[((size_t)b * 32 + ka) * 4096 + hw];
    int kb = hid * 16 + j;
    bv[j] = (_Float16)fw[kb * 128 + n];
  }
  acc = WMMA_F16(av, bv, acc);
  for (int r = 0; r < 8; ++r) {
    int pr = Mt * 16 + hid * 8 + r, nc = Nt * 16 + m;
    float v = gelu_exact(acc[r] + fb[nc]);
    t1[(size_t)pr * 128 + nc] = (_Float16)v;
  }
}

// ---- GEMM2 (WMMA f16): t1 (16384x128) @ fi1w (128x1024) + b, scatter into t ----
__global__ void k_gemm2(const _Float16* __restrict__ t1, const float* __restrict__ fw,
                        const float* __restrict__ fb, float* __restrict__ t) {
  int Mt = blockIdx.x, Nt = blockIdx.y;
  int lane = threadIdx.x & 31, m = lane & 15, hid = lane >> 4;
  int p = Mt * 16 + m, n = Nt * 16 + m;
  v8f acc = {};
  for (int kc = 0; kc < 4; ++kc) {
    int base = kc * 32;
    if (kc < 3) __builtin_prefetch(&fw[(size_t)(base + 32) * 1024 + n], 0, 0);
    v16h av, bv;
    for (int j = 0; j < 16; ++j) {
      int ka = base + hid * 16 + j;
      av[j] = t1[(size_t)p * 128 + ka];
      int kb = base + hid * 16 + j;
      bv[j] = (_Float16)fw[(size_t)kb * 1024 + n];
    }
    acc = WMMA_F16(av, bv, acc);
  }
  for (int r = 0; r < 8; ++r) {
    int pr = Mt * 16 + hid * 8 + r, nc = Nt * 16 + m;
    int b = pr >> 12, hw = pr & 4095, cc = nc >> 5, ss = nc & 31;
    t[(((size_t)b * 32 + ss) * 32 + cc) * 4096 + hw] = acc[r] + fb[nc];
  }
}

// ---- Hermitian completion from planar Ym -> interleaved uftf (img, x, y<31) ----
__global__ void k_herm(const float* __restrict__ Ymp, float* __restrict__ uftf) {
  int id = blockIdx.x * blockDim.x + threadIdx.x;
  if (id >= 4096 * 32 * 31) return;
  int yy = id % 31, rest = id / 31, x = rest & 31, img = rest >> 5;
  float r, i;
  if (yy < 16) {
    size_t s = (size_t)img * 1024 + x * 16 + yy;
    r = Ymp[s]; i = Ymp[s + 512];
  } else {
    int sx = 31 - x, sy = 31 - yy;
    size_t s = (size_t)img * 1024 + sx * 16 + sy;
    r = Ymp[s]; i = -Ymp[s + 512];
  }
  size_t d = (((size_t)img * 32 + x) * 31 + yy) * 2;
  uftf[d] = r; uftf[d + 1] = i;
}

// ---- Non-uniform inverse transform (WMMA f16, separable basis in LDS) ----
__global__ void k_point(const float* __restrict__ uftf, const float* __restrict__ xy,
                        const float* __restrict__ bw, const float* __restrict__ bb,
                        float* __restrict__ pty) {
  int bs = blockIdx.x, nt = blockIdx.y;
  __shared__ float sE1r[16][32], sE1i[16][32], sE2r[16][32], sE2i[16][32];
  __shared__ float sXY[16][2];
  int tid = threadIdx.x;
  if (tid < 32) {
    int n = tid >> 1, c = tid & 1;
    sXY[n][c] = xy[((size_t)bs * 512 + nt * 16 + n) * 2 + c];
  }
  __syncthreads();
  for (int idx = tid; idx < 16 * 63; idx += 64) {
    int n = idx / 63, k = idx % 63;
    float s, c;
    if (k < 32) {
      int k1 = (k < 16) ? k : k - 32;
      float u = sXY[n][0] * (float)k1; u -= floorf(u);
      __sincosf(6.28318530717958647f * u, &s, &c);
      sE1r[n][k] = c; sE1i[n][k] = s;
    } else {
      int y = k - 32;
      int k2 = (y < 16) ? y : y - 31;
      float u = sXY[n][1] * (float)k2; u -= floorf(u);
      __sincosf(6.28318530717958647f * u, &s, &c);
      sE2r[n][y] = c; sE2i[n][y] = s;
    }
  }
  __syncthreads();

  int lane = tid & 31, wv = tid >> 5;
  int m = lane & 15, hid = lane >> 4, ct = wv;
  v8f acc = {};
  const float* ufb = uftf + ((size_t)(bs * 32 + ct * 16 + m) * 32) * 31 * 2;
  for (int x = 0; x < 32; ++x) {
    float e1r = sE1r[m][x], e1i = sE1i[m][x];
    const float* ufr = ufb + (size_t)x * 31 * 2;
    for (int q = 0; q < 2; ++q) {
      v16h av, bv;
      for (int j = 0; j < 16; ++j) {
        int kb = q * 32 + hid * 16 + j, yb = kb >> 1;
        float r2 = (yb < 31) ? sE2r[m][yb] : 0.f;
        float i2 = (yb < 31) ? sE2i[m][yb] : 0.f;
        float bvv = (kb & 1) ? (e1r * i2 + e1i * r2) : (e1r * r2 - e1i * i2);
        bv[j] = (_Float16)bvv;
        int ka = q * 32 + hid * 16 + j, ya = ka >> 1;
        float ur = (ya < 31) ? ufr[ya * 2] : 0.f;
        float ui = (ya < 31) ? ufr[ya * 2 + 1] : 0.f;
        float avv = (ka & 1) ? (-ui) : ur;
        av[j] = (_Float16)avv;
      }
      acc = WMMA_F16(av, bv, acc);
    }
  }
  float xn = sXY[m][0], yn = sXY[m][1];
  for (int r = 0; r < 8; ++r) {
    int c = ct * 16 + hid * 8 + r, ng = nt * 16 + m;
    float v = acc[r] + bw[c * 2] * xn + bw[c * 2 + 1] * yn + bb[c];
    pty[((size_t)bs * 32 + c) * 512 + ng] = v;
  }
}

// ---- GEMM3 (WMMA f16): pty gathered (65536x32) @ fc1w (32x128), gelu -> z1 f16 ----
__global__ void k_gemm3(const float* __restrict__ pty, const float* __restrict__ fw,
                        const float* __restrict__ fb, _Float16* __restrict__ z1) {
  int Mt = blockIdx.x, Nt = blockIdx.y;
  int lane = threadIdx.x & 31, m = lane & 15, hid = lane >> 4;
  int q = Mt * 16 + m, bsq = q >> 9, nq = q & 511;
  int n = Nt * 16 + m;
  v16h av, bv; v8f acc = {};
  for (int j = 0; j < 16; ++j) {
    int ka = hid * 16 + j;
    av[j] = (_Float16)pty[((size_t)bsq * 32 + ka) * 512 + nq];
    int kb = hid * 16 + j;
    bv[j] = (_Float16)fw[kb * 128 + n];
  }
  acc = WMMA_F16(av, bv, acc);
  for (int r = 0; r < 8; ++r) {
    int qr = Mt * 16 + hid * 8 + r, nc = Nt * 16 + m;
    float v = gelu_exact(acc[r] + fb[nc]);
    z1[(size_t)qr * 128 + nc] = (_Float16)v;
  }
}

// ---- fc2: (65536x128) @ (128x1) + b -> out ----
__global__ void k_fc2(const _Float16* __restrict__ z1, const float* __restrict__ fw,
                      const float* __restrict__ fb, float* __restrict__ out) {
  int q = blockIdx.x * blockDim.x + threadIdx.x;
  if (q >= 65536) return;
  const _Float16* zr = z1 + (size_t)q * 128;
  float acc = fb[0];
  for (int j = 0; j < 128; ++j) acc += (float)zr[j] * fw[j];
  out[q] = acc;
}

// ============================================================
extern "C" void kernel_launch(void* const* d_in, const int* in_sizes, int n_in,
                              void* d_out, int out_size, void* d_ws, size_t ws_size,
                              hipStream_t stream) {
  (void)in_sizes; (void)n_in; (void)out_size; (void)ws_size;
  const float* u    = (const float*)d_in[0];
  const float* xy   = (const float*)d_in[1];
  const float* sw1r = (const float*)d_in[2];
  const float* sw1i = (const float*)d_in[3];
  const float* sw2r = (const float*)d_in[4];
  const float* sw2i = (const float*)d_in[5];
  const float* cw   = (const float*)d_in[6];
  const float* cb   = (const float*)d_in[7];
  const float* bw   = (const float*)d_in[8];
  const float* bb   = (const float*)d_in[9];
  const float* fc0w = (const float*)d_in[10];
  const float* fc0b = (const float*)d_in[11];
  const float* fi0w = (const float*)d_in[12];
  const float* fi0b = (const float*)d_in[13];
  const float* fi1w = (const float*)d_in[14];
  const float* fi1b = (const float*)d_in[15];
  const float* fc1w = (const float*)d_in[16];
  const float* fc1b = (const float*)d_in[17];
  const float* fc2w = (const float*)d_in[18];
  const float* fc2b = (const float*)d_in[19];
  float* out = (float*)d_out;

  char* W = (char*)d_ws;
  size_t o = 0;
  auto take = [&](size_t nbytes) -> char* {
    char* p = W + o;
    o += (nbytes + 255) & ~(size_t)255;
    return p;
  };
  float*     EA   = (float*)take(4096 * 4);
  float*     EE   = (float*)take(4096 * 4);
  float*     BBt  = (float*)take(4096 * 4);
  float*     BDt  = (float*)take(4096 * 4);
  float*     xg   = (float*)take((size_t)4 * 32 * 4096 * 4);
  float*     t    = (float*)take((size_t)128 * 32 * 4096 * 4);
  float*     tmpA = (float*)take((size_t)4096 * 4096 * 4);   // planar re/im
  float*     ybuf = (float*)take((size_t)128 * 32 * 4096 * 4);
  float*     Xm   = (float*)take((size_t)4096 * 1024 * 4);   // planar
  float*     Ym   = (float*)take((size_t)4096 * 1024 * 4);   // planar
  float*     uftf = (float*)take((size_t)4096 * 32 * 31 * 2 * 4);
  float*     pty  = (float*)take((size_t)128 * 32 * 512 * 4);
  _Float16*  t1   = (_Float16*)take((size_t)16384 * 128 * 2);
  _Float16*  z1   = (_Float16*)take((size_t)65536 * 128 * 2);

  k_tables<<<64, 256, 0, stream>>>(EA, EE, BBt, BDt);
  k_fc0<<<64, 256, 0, stream>>>(u, fc0w, fc0b, xg);

  // spectral layers 0..3 on xg (batch 4, nimg = 128)
  for (int l = 0; l < 4; ++l) {
    k_wA<<<dim3(128, 4, 4), 32, 0, stream>>>(xg, EA, tmpA);
    k_wB<<<dim3(128, 2, 2), 32, 0, stream>>>(tmpA, BBt, Xm);
    k_mixw<<<dim3(512, 1, 4), 32, 0, stream>>>(Xm, sw1r, sw1i, sw2r, sw2i, Ym, l, 4);
    k_wD<<<dim3(128, 2, 8), 32, 0, stream>>>(Ym, BDt, tmpA);
    k_wE<<<dim3(128, 4, 4), 32, 0, stream>>>(EE, tmpA, ybuf);
    k_convact<<<64, 256, 0, stream>>>(ybuf, xg, cw, cb, l, 4);
  }

  // interp head: xg -> t1 -> t (bs=128, c=32, h, w)
  k_gemm1<<<dim3(1024, 8), 32, 0, stream>>>(xg, fi0w, fi0b, t1);
  k_gemm2<<<dim3(1024, 64), 32, 0, stream>>>(t1, fi1w, fi1b, t);

  // layer 4 on t (batch 128, nimg = 4096)
  k_wA<<<dim3(4096, 4, 4), 32, 0, stream>>>(t, EA, tmpA);
  k_wB<<<dim3(4096, 2, 2), 32, 0, stream>>>(tmpA, BBt, Xm);
  k_mixw<<<dim3(512, 8, 4), 32, 0, stream>>>(Xm, sw1r, sw1i, sw2r, sw2i, Ym, 4, 128);
  k_wD<<<dim3(4096, 2, 8), 32, 0, stream>>>(Ym, BDt, tmpA);
  k_wE<<<dim3(4096, 4, 4), 32, 0, stream>>>(EE, tmpA, ybuf);
  k_convact<<<2048, 256, 0, stream>>>(ybuf, t, cw, cb, 4, 128);

  // layer 5: forward transform + mix, hermitian completion, point evaluation
  k_wA<<<dim3(4096, 4, 4), 32, 0, stream>>>(t, EA, tmpA);
  k_wB<<<dim3(4096, 2, 2), 32, 0, stream>>>(tmpA, BBt, Xm);
  k_mixw<<<dim3(512, 8, 4), 32, 0, stream>>>(Xm, sw1r, sw1i, sw2r, sw2i, Ym, 5, 128);
  k_herm<<<(4096 * 32 * 31 + 255) / 256, 256, 0, stream>>>(Ym, uftf);
  k_point<<<dim3(128, 32), 64, 0, stream>>>(uftf, xy, bw, bb, pty);

  // output head
  k_gemm3<<<dim3(4096, 8), 32, 0, stream>>>(pty, fc1w, fc1b, z1);
  k_fc2<<<256, 256, 0, stream>>>(z1, fc2w, fc2b, out);
}
